// GraphAttentionHeadModule_50096498540727
// MI455X (gfx1250) — matrix operand order repaired and checked by reading, not compiled, not run on hardware
//
#include <hip/hip_runtime.h>
#include <hip/hip_bf16.h>
#include <math.h>

typedef __attribute__((ext_vector_type(16))) _Float16 v16h;
typedef __attribute__((ext_vector_type(8)))  float    v8f;

#define N_NODES 8192
#define IN_FEAT 256
#define OUT_FEAT 64
#define NSPLIT 4
#define JTILE 32
#define GAT_ALPHA 0.2f
#define GAT_NEG_BIG -9.0e15f

// ---------------------------------------------------------------------------
// Kernel 1: h' = X @ W  (f32), write h'^T as f16 [64][8192], plus s1/s2.
// ---------------------------------------------------------------------------
__global__ __launch_bounds__(256) void k_hprime(const float* __restrict__ X,
                                                const float* __restrict__ W,
                                                const float* __restrict__ a1,
                                                const float* __restrict__ a2,
                                                _Float16* __restrict__ hpT,
                                                float* __restrict__ s1,
                                                float* __restrict__ s2) {
    const int t    = threadIdx.x;
    const int feat = t & 63;          // 0..63
    const int rl   = t >> 6;          // 0..3 local row
    const int row  = blockIdx.x * 4 + rl;

    const float* xr = X + (size_t)row * IN_FEAT;
    float acc = 0.0f;
#pragma unroll 8
    for (int k = 0; k < IN_FEAT; ++k)
        acc = fmaf(xr[k], W[k * OUT_FEAT + feat], acc);

    hpT[(size_t)feat * N_NODES + row] = (_Float16)acc;

    __shared__ float r1[256];
    __shared__ float r2[256];
    r1[t] = acc * a1[feat];
    r2[t] = acc * a2[feat];
    __syncthreads();
    if (feat == 0) {
        float u = 0.0f, v = 0.0f;
        for (int i = 0; i < 64; ++i) { u += r1[rl * 64 + i]; v += r2[rl * 64 + i]; }
        s1[row] = u;
        s2[row] = v;
    }
}

// ---------------------------------------------------------------------------
// Kernel 1b: gmax = max_j s2[j]  (single block reduction over 8192 floats).
// leakyrelu is monotone => M_i = leakyrelu(s1_i + gmax) upper-bounds every
// score of row i, giving a softmax shift that needs no online-max machinery.
// ---------------------------------------------------------------------------
__global__ __launch_bounds__(256) void k_s2max(const float* __restrict__ s2,
                                               float* __restrict__ gmax) {
    const int t = threadIdx.x;
    float m = -INFINITY;
    for (int i = t; i < N_NODES; i += 256) m = fmaxf(m, s2[i]);
    __shared__ float red[256];
    red[t] = m;
    __syncthreads();
    for (int w = 128; w > 0; w >>= 1) {
        if (t < w) red[t] = fmaxf(red[t], red[t + w]);
        __syncthreads();
    }
    if (t == 0) *gmax = red[0];
}

// ---------------------------------------------------------------------------
// Kernel 2: fused masked-softmax + P@V, WMMA f32<=f16 16x16x32.
// One wave (32 threads) per 16-row tile per column split. No cross-lane ops
// in the streaming loop: P is exp(e - M_i) with the fixed bound M_i, built
// directly in the 16-bit A-matrix VGPR layout:
//   lane l owns row (l&15), K = (l>>4)*8 + {0..7} in halves 0..7, +16 in 8..15.
// B (32x16 f16): lane l owns column (l&15), K = (l>>4)*8 + {0..7}, +16 in 8..15 -> two
//   16B runs of hpT[feat][node].
// C/D: lane l, VGPR r -> row r + 8*(l>>4), col (l&15).
// ---------------------------------------------------------------------------
__global__ __launch_bounds__(32) void k_attn(const int* __restrict__ adj,
                                             const _Float16* __restrict__ hpT,
                                             const float* __restrict__ s1,
                                             const float* __restrict__ s2,
                                             const float* __restrict__ gmaxp,
                                             float* __restrict__ lP,
                                             float* __restrict__ accP) {
    const int lane  = threadIdx.x;
    const int lo    = lane & 15;
    const int hi    = lane >> 4;
    const int k0    = hi * 8;
    const int i0    = blockIdx.x * 16;
    const int split = blockIdx.y;
    const int jlen  = N_NODES / NSPLIT;
    const int jstart = split * jlen;
    const int jend   = jstart + jlen;

    const int   rowg   = i0 + lo;
    const float s1v    = s1[rowg];
    const int*  adjRow = adj + (size_t)rowg * N_NODES;

    // fixed softmax shift: upper bound of every score in this row
    const float gmax = *gmaxp;
    float M = s1v + gmax;
    M = fmaxf(M, GAT_ALPHA * M);

    float tsum = 0.0f;               // per-lane partial of row (l&15) exp-sum
    const v8f vzero = {0.f, 0.f, 0.f, 0.f, 0.f, 0.f, 0.f, 0.f};
    v8f acc[4];
#pragma unroll
    for (int t = 0; t < 4; ++t) acc[t] = vzero;

    for (int j0 = jstart; j0 < jend; j0 += JTILE) {
        if (j0 + JTILE < jend)
            __builtin_prefetch(adjRow + j0 + JTILE + k0, 0, 0);

        // ---- the 16 (row, k) score inputs this lane owns ----
        const float4 s2a = *(const float4*)(s2 + j0 + k0);
        const float4 s2b = *(const float4*)(s2 + j0 + k0 + 4);
        const float4 s2c = *(const float4*)(s2 + j0 + k0 + 16);
        const float4 s2d = *(const float4*)(s2 + j0 + k0 + 20);
        const int4  ada = *(const int4*)(adjRow + j0 + k0);
        const int4  adb = *(const int4*)(adjRow + j0 + k0 + 4);
        const int4  adc = *(const int4*)(adjRow + j0 + k0 + 16);
        const int4  add = *(const int4*)(adjRow + j0 + k0 + 20);

        const float s2f[16] = {s2a.x, s2a.y, s2a.z, s2a.w, s2b.x, s2b.y, s2b.z, s2b.w,
                               s2c.x, s2c.y, s2c.z, s2c.w, s2d.x, s2d.y, s2d.z, s2d.w};
        const int   adf[16] = {ada.x, ada.y, ada.z, ada.w, adb.x, adb.y, adb.z, adb.w,
                               adc.x, adc.y, adc.z, adc.w, add.x, add.y, add.z, add.w};

        // ---- scores -> probabilities, straight into the A fragment ----
        v16h af;
#pragma unroll
        for (int q = 0; q < 16; ++q) {
            float e = s1v + s2f[q];
            e = fmaxf(e, GAT_ALPHA * e);              // leaky_relu, alpha in (0,1)
            e = (adf[q] > 0) ? e : GAT_NEG_BIG;       // mask -> exp underflows to 0
            const float pq = __expf(e - M);           // in (0, 1]
            const _Float16 ph = (_Float16)pq;
            af[q] = ph;
            tsum += (float)ph;                        // denominator matches f16 P
        }

        // ---- B fragments from hpT (f16, [feat][node]) + WMMA ----
#pragma unroll
        for (int t = 0; t < 4; ++t) {
            const _Float16* bp = hpT + (size_t)(t * 16 + lo) * N_NODES + j0 + hi * 8;
            union { v16h v; float4 f[2]; } bu;
            bu.f[0] = *(const float4*)(bp);
            bu.f[1] = *(const float4*)(bp + 16);
            acc[t] = __builtin_amdgcn_wmma_f32_16x16x32_f16(
                false, af, false, bu.v, (short)0, acc[t], false, false);
        }
    }

    // ---- fold the two half-wave partials of each row's exp-sum ----
    const float rsum = tsum + __shfl_xor(tsum, 16, 32);
    if (hi == 0)
        lP[(size_t)split * N_NODES + rowg] = rsum;

#pragma unroll
    for (int t = 0; t < 4; ++t)
#pragma unroll
        for (int r = 0; r < 8; ++r) {
            const int row = r + 8 * hi;
            accP[((size_t)split * N_NODES + i0 + row) * OUT_FEAT + t * 16 + lo] = acc[t][r];
        }
}

// ---------------------------------------------------------------------------
// Kernel 3: merge column splits (same shift M_i everywhere -> plain sums),
// normalize, ELU.
// ---------------------------------------------------------------------------
__global__ __launch_bounds__(256) void k_combine(const float* __restrict__ lP,
                                                 const float* __restrict__ accP,
                                                 float* __restrict__ out) {
    const int idx = blockIdx.x * 256 + threadIdx.x;
    if (idx >= N_NODES * OUT_FEAT) return;
    const int row = idx >> 6;
    const int col = idx & 63;

    float denom = 0.0f, num = 0.0f;
#pragma unroll
    for (int s = 0; s < NSPLIT; ++s) {
        denom += lP[(size_t)s * N_NODES + row];
        num   += accP[((size_t)s * N_NODES + row) * OUT_FEAT + col];
    }
    const float h = num / denom;
    out[idx] = (h > 0.0f) ? h : (__expf(h) - 1.0f);   // ELU
}

// ---------------------------------------------------------------------------
extern "C" void kernel_launch(void* const* d_in, const int* in_sizes, int n_in,
                              void* d_out, int out_size, void* d_ws, size_t ws_size,
                              hipStream_t stream) {
    const float* X  = (const float*)d_in[0];   // node_features [8192,256]
    const int*   A  = (const int*)d_in[1];     // adj_matrix    [8192,8192]
    const float* W  = (const float*)d_in[2];   // W             [256,64]
    const float* a1 = (const float*)d_in[3];   // a_1           [64,1]
    const float* a2 = (const float*)d_in[4];   // a_2           [64,1]
    float* out = (float*)d_out;

    // workspace layout
    char* ws = (char*)d_ws;
    _Float16* hpT = (_Float16*)ws;                         // 64*8192*2   = 1 MB
    float* s1   = (float*)(ws + (1 << 20));                // 32 KB
    float* s2   = s1 + N_NODES;                            // 32 KB
    float* gmax = s2 + N_NODES;                            // 1 float (+pad)
    float* lP   = gmax + 64;                               // NSPLIT*8192*4
    float* accP = lP + (size_t)NSPLIT * N_NODES;           // NSPLIT*8192*64*4 = 8 MB

    k_hprime<<<N_NODES / 4, 256, 0, stream>>>(X, W, a1, a2, hpT, s1, s2);
    k_s2max<<<1, 256, 0, stream>>>(s2, gmax);

    dim3 g2(N_NODES / 16, NSPLIT);
    k_attn<<<g2, 32, 0, stream>>>(A, hpT, s1, s2, gmax, lP, accP);

    k_combine<<<(N_NODES * OUT_FEAT + 255) / 256, 256, 0, stream>>>(lP, accP, out);
}
